// SelfAttentionBlock_25804163515033
// MI455X (gfx1250) — compile-verified, matrix operand order checked by reading, not run on hardware
//
#include <hip/hip_runtime.h>

#define E     1024
#define HEADS 16
#define HD    64
#define SEQ   2048
#define BATCH 4
#define TFEAT 128
#define MROWS (BATCH * SEQ)          // 8192
#define KT    (E / 32)               // 32 k-tiles
#define NT    (E / 16)               // 64 n-tiles
#define MT    (MROWS / 16)           // 512 m-tiles

typedef __attribute__((ext_vector_type(16))) __bf16 bf16x16;
typedef __attribute__((ext_vector_type(8)))  float  f32x8;

static __device__ inline f32x8 wmma_bf16(bf16x16 a, bf16x16 b, f32x8 c) {
  return __builtin_amdgcn_wmma_f32_16x16x32_bf16(false, a, false, b, (short)0, c,
                                                 false, false);
}

// bf16x3 split: f ~= (float)h + (float)l
static __device__ inline void splitf(float f, __bf16 &h, __bf16 &l) {
  h = (__bf16)f;
  l = (__bf16)(f - (float)h);
}

// C += Ah*Bh + Ah*Bl + Al*Bh
static __device__ inline f32x8 mma3(bf16x16 ah, bf16x16 al,
                                    bf16x16 bh, bf16x16 bl, f32x8 c) {
  c = wmma_bf16(ah, bh, c);
  c = wmma_bf16(ah, bl, c);
  c = wmma_bf16(al, bh, c);
  return c;
}

// ---------------- kernel 1: time-dependent head weights ----------------
__global__ void tw_kernel(const float* __restrict__ time_enc,
                          const float* __restrict__ Wt,
                          const float* __restrict__ bt,
                          float* __restrict__ tw) {
  int b = blockIdx.x;
  int lane = threadIdx.x;
  int h = lane & 15;
  float acc = 0.f;
  for (int t = 0; t < TFEAT; ++t)
    acc += time_enc[b * TFEAT + t] * Wt[t * HEADS + h];
  acc += bt[h];
  float mx = acc;
  for (int m = 1; m < 16; m <<= 1) mx = fmaxf(mx, __shfl_xor(mx, m, 32));
  float e = __expf(acc - mx);
  float sum = e;
  for (int m = 1; m < 16; m <<= 1) sum += __shfl_xor(sum, m, 32);
  if (lane < 16) tw[b * HEADS + h] = e / sum;
}

// ------------- A-fragment pre-split: src [MROWS x E] f32 -> hi/lo frags -------------
// Fragment layout: [mt][kt][lane][16] __bf16, value = src[mt*16 + (lane&15)][kt*32 + (lane>>4)*16 + i]
__global__ void __launch_bounds__(256) afrag_kernel(const float* __restrict__ src,
    __bf16* __restrict__ fh, __bf16* __restrict__ fl) {
  int wid = blockIdx.x * 8 + (threadIdx.x >> 5);     // MT*KT waves
  int kt = wid & (KT - 1);
  int mt = wid >> 5;
  int lane = threadIdx.x & 31;
  const float* p = src + (size_t)(mt * 16 + (lane & 15)) * E + kt * 32 + ((lane >> 4) << 4);
  size_t o = (((size_t)mt * KT + kt) * 32 + lane) * 16;
  #pragma unroll
  for (int i = 0; i < 16; ++i) {
    __bf16 h, l; splitf(p[i], h, l);
    fh[o + i] = h; fl[o + i] = l;
  }
}

// ------------- B-fragment pre-split: W [E x E] f32 -> hi/lo frags -------------
// Fragment layout: [kt][nt][lane][16], value = W[kt*32 + (lane>>4)*16 + i][nt*16 + (lane&15)]
__global__ void __launch_bounds__(256) bfrag_kernel(const float* __restrict__ W,
    __bf16* __restrict__ fh, __bf16* __restrict__ fl) {
  int wid = blockIdx.x * 8 + (threadIdx.x >> 5);     // KT*NT waves
  int nt = wid & (NT - 1);
  int kt = wid >> 6;
  int lane = threadIdx.x & 31;
  const float* p = W + (size_t)(kt * 32 + ((lane >> 4) << 4)) * E + nt * 16 + (lane & 15);
  size_t o = (((size_t)kt * NT + nt) * 32 + lane) * 16;
  #pragma unroll
  for (int i = 0; i < 16; ++i) {
    __bf16 h, l; splitf(p[(size_t)i * E], h, l);
    fh[o + i] = h; fl[o + i] = l;
  }
}

// ---------------- kernel 2: fused QKV projection from fragments ----------------
// Outputs: Qh/Ql, Kh/Kl as bf16 [B,H,S,D]; Vth/Vtl as bf16 [B,H,D,S]
__global__ void __launch_bounds__(256) qkv_kernel(
    const __bf16* __restrict__ xfh, const __bf16* __restrict__ xfl,
    const __bf16* __restrict__ wqh, const __bf16* __restrict__ wql,
    const __bf16* __restrict__ wkh, const __bf16* __restrict__ wkl,
    const __bf16* __restrict__ wvh, const __bf16* __restrict__ wvl,
    __bf16* __restrict__ Qh, __bf16* __restrict__ Ql,
    __bf16* __restrict__ Kh, __bf16* __restrict__ Kl,
    __bf16* __restrict__ Vth, __bf16* __restrict__ Vtl) {
  int wid = blockIdx.x * 8 + (threadIdx.x >> 5);
  int nt = wid & (NT - 1);
  int mt = wid >> 6;
  int lane = threadIdx.x & 31;
  f32x8 cq = {}; f32x8 ck = {}; f32x8 cv = {};
  for (int kt = 0; kt < KT; ++kt) {
    size_t ao = (((size_t)mt * KT + kt) * 32 + lane) * 16;
    size_t bo = (((size_t)kt * NT + nt) * 32 + lane) * 16;
    bf16x16 ah = *(const bf16x16*)(xfh + ao);
    bf16x16 al = *(const bf16x16*)(xfl + ao);
    cq = mma3(ah, al, *(const bf16x16*)(wqh + bo), *(const bf16x16*)(wql + bo), cq);
    ck = mma3(ah, al, *(const bf16x16*)(wkh + bo), *(const bf16x16*)(wkl + bo), ck);
    cv = mma3(ah, al, *(const bf16x16*)(wvh + bo), *(const bf16x16*)(wvl + bo), cv);
  }
  // store split results
  int col = nt * 16 + (lane & 15);
  int h = col >> 6, d = col & 63;
  int row0 = mt * 16 + ((lane >> 4) << 3);
  #pragma unroll
  for (int r = 0; r < 8; ++r) {
    int row = row0 + r;
    int b = row >> 11, s = row & (SEQ - 1);
    int hb = b * HEADS + h;
    size_t qi = ((size_t)hb * SEQ + s) * HD + d;     // [B,H,S,D]
    size_t vi = ((size_t)hb * HD + d) * SEQ + s;     // [B,H,D,S]
    __bf16 hh, ll;
    splitf(cq[r], hh, ll); Qh[qi] = hh; Ql[qi] = ll;
    splitf(ck[r], hh, ll); Kh[qi] = hh; Kl[qi] = ll;
    splitf(cv[r], hh, ll); Vth[vi] = hh; Vtl[vi] = ll;
  }
}

// ---------------- kernel 3: flash attention, all operands pre-split bf16 ----------------
__global__ void __launch_bounds__(256) attn_kernel(
    const __bf16* __restrict__ Qh, const __bf16* __restrict__ Ql,
    const __bf16* __restrict__ Kh, const __bf16* __restrict__ Kl,
    const __bf16* __restrict__ Vth, const __bf16* __restrict__ Vtl,
    const float* __restrict__ tw, float* __restrict__ ctx) {
  __shared__ float pbuf[8][16 * 32];
  int widl = threadIdx.x >> 5;
  int wid  = blockIdx.x * 8 + widl;
  int qb   = wid & 127;
  int h    = (wid >> 7) & 15;
  int b    = wid >> 11;
  int lane = threadIdx.x & 31;
  int lr = lane & 15;
  int lh = (lane >> 4) << 4;
  int hb = b * HEADS + h;

  const __bf16* qhp = Qh + ((size_t)hb * SEQ + qb * 16) * HD;
  const __bf16* qlp = Ql + ((size_t)hb * SEQ + qb * 16) * HD;
  const __bf16* khp = Kh + (size_t)hb * SEQ * HD;
  const __bf16* klp = Kl + (size_t)hb * SEQ * HD;
  const __bf16* vhp = Vth + (size_t)hb * HD * SEQ;
  const __bf16* vlp = Vtl + (size_t)hb * HD * SEQ;
  float scl = 0.125f * tw[hb];

  size_t qo = (size_t)lr * HD + lh;
  bf16x16 q0h = *(const bf16x16*)(qhp + qo);
  bf16x16 q0l = *(const bf16x16*)(qlp + qo);
  bf16x16 q1h = *(const bf16x16*)(qhp + qo + 32);
  bf16x16 q1l = *(const bf16x16*)(qlp + qo + 32);

  f32x8 o0 = {}, o1 = {}, o2 = {}, o3 = {};
  float mrow[8], lrow[8];
  #pragma unroll
  for (int r = 0; r < 8; ++r) { mrow[r] = -1e30f; lrow[r] = 0.f; }
  float* myp = pbuf[widl];

  for (int kb = 0; kb < SEQ; kb += 32) {
    // ---- scores: keys kb..+15 (s0), kb+16..+31 (s1); K rows are WMMA B-operand chunks ----
    f32x8 s0 = {}, s1 = {};
    {
      size_t k0 = (size_t)(kb + lr) * HD + lh;        // group 0
      s0 = mma3(q0h, q0l, *(const bf16x16*)(khp + k0),
                          *(const bf16x16*)(klp + k0), s0);
      s0 = mma3(q1h, q1l, *(const bf16x16*)(khp + k0 + 32),
                          *(const bf16x16*)(klp + k0 + 32), s0);
      size_t k1 = (size_t)(kb + 16 + lr) * HD + lh;   // group 1
      s1 = mma3(q0h, q0l, *(const bf16x16*)(khp + k1),
                          *(const bf16x16*)(klp + k1), s1);
      s1 = mma3(q1h, q1l, *(const bf16x16*)(khp + k1 + 32),
                          *(const bf16x16*)(klp + k1 + 32), s1);
    }
    // ---- online softmax update ----
    float p0[8], p1[8], alpha[8];
    #pragma unroll
    for (int r = 0; r < 8; ++r) {
      float a0 = s0[r] * scl, a1 = s1[r] * scl;
      float mx = fmaxf(a0, a1);
      for (int m = 1; m < 16; m <<= 1) mx = fmaxf(mx, __shfl_xor(mx, m, 32));
      float mnew = fmaxf(mrow[r], mx);
      float al_  = __expf(mrow[r] - mnew);
      p0[r] = __expf(a0 - mnew);
      p1[r] = __expf(a1 - mnew);
      float rs = p0[r] + p1[r];
      for (int m = 1; m < 16; m <<= 1) rs += __shfl_xor(rs, m, 32);
      lrow[r] = lrow[r] * al_ + rs;
      mrow[r] = mnew;
      alpha[r] = al_;
    }
    #pragma unroll
    for (int r = 0; r < 8; ++r) {
      o0[r] *= alpha[r]; o1[r] *= alpha[r];
      o2[r] *= alpha[r]; o3[r] *= alpha[r];
    }
    // ---- P: C layout -> A layout via per-wave LDS slice, then split ----
    {
      int row = (lane >> 4) << 3;
      #pragma unroll
      for (int r = 0; r < 8; ++r) {
        myp[(row + r) * 32 + lr]      = p0[r];
        myp[(row + r) * 32 + 16 + lr] = p1[r];
      }
    }
    bf16x16 ph, pl;
    #pragma unroll
    for (int i = 0; i < 16; ++i) {
      __bf16 hh, ll; splitf(myp[lr * 32 + lh + i], hh, ll);
      ph[i] = hh; pl[i] = ll;
    }
    // ---- O += P @ V, V^T gives contiguous B-operand chunks ----
    size_t v0 = (size_t)lr * SEQ + kb + lh;
    o0 = mma3(ph, pl, *(const bf16x16*)(vhp + v0),
                      *(const bf16x16*)(vlp + v0), o0);
    size_t v1 = v0 + (size_t)16 * SEQ;
    o1 = mma3(ph, pl, *(const bf16x16*)(vhp + v1),
                      *(const bf16x16*)(vlp + v1), o1);
    size_t v2 = v1 + (size_t)16 * SEQ;
    o2 = mma3(ph, pl, *(const bf16x16*)(vhp + v2),
                      *(const bf16x16*)(vlp + v2), o2);
    size_t v3 = v2 + (size_t)16 * SEQ;
    o3 = mma3(ph, pl, *(const bf16x16*)(vhp + v3),
                      *(const bf16x16*)(vlp + v3), o3);
  }
  // ---- normalize and store into ctx [B,S,E] f32 ----
  #pragma unroll
  for (int r = 0; r < 8; ++r) {
    float inv = 1.0f / lrow[r];
    o0[r] *= inv; o1[r] *= inv; o2[r] *= inv; o3[r] *= inv;
  }
  int row0 = qb * 16 + ((lane >> 4) << 3);
  float* outp = ctx + (size_t)b * SEQ * E;
  #pragma unroll
  for (int r = 0; r < 8; ++r) {
    size_t base = (size_t)(row0 + r) * E + h * HD + lr;
    outp[base]      = o0[r];
    outp[base + 16] = o1[r];
    outp[base + 32] = o2[r];
    outp[base + 48] = o3[r];
  }
}

// ---------------- kernel 4: output projection + bias, from fragments ----------------
__global__ void __launch_bounds__(256) oproj_kernel(
    const __bf16* __restrict__ cfh, const __bf16* __restrict__ cfl,
    const __bf16* __restrict__ woh, const __bf16* __restrict__ wol,
    const float* __restrict__ bo, float* __restrict__ out) {
  int wid = blockIdx.x * 8 + (threadIdx.x >> 5);
  int nt = wid & (NT - 1);
  int mt = wid >> 6;
  int lane = threadIdx.x & 31;
  f32x8 c = {};
  for (int kt = 0; kt < KT; ++kt) {
    size_t ao = (((size_t)mt * KT + kt) * 32 + lane) * 16;
    size_t bo_ = (((size_t)kt * NT + nt) * 32 + lane) * 16;
    c = mma3(*(const bf16x16*)(cfh + ao), *(const bf16x16*)(cfl + ao),
             *(const bf16x16*)(woh + bo_), *(const bf16x16*)(wol + bo_), c);
  }
  int col = nt * 16 + (lane & 15);
  float bias = bo[col];
  int row0 = mt * 16 + ((lane >> 4) << 3);
  #pragma unroll
  for (int r = 0; r < 8; ++r)
    out[(size_t)(row0 + r) * E + col] = c[r] + bias;
}

extern "C" void kernel_launch(void* const* d_in, const int* in_sizes, int n_in,
                              void* d_out, int out_size, void* d_ws, size_t ws_size,
                              hipStream_t stream) {
  const float* x        = (const float*)d_in[0];
  const float* time_enc = (const float*)d_in[1];
  const float* Wq       = (const float*)d_in[2];
  const float* Wk       = (const float*)d_in[3];
  const float* Wv       = (const float*)d_in[4];
  const float* Wo       = (const float*)d_in[5];
  const float* bo       = (const float*)d_in[6];
  const float* Wt       = (const float*)d_in[7];
  const float* bt       = (const float*)d_in[8];
  float* out = (float*)d_out;

  const size_t AFRAG = (size_t)MROWS * E;   // 8M elems
  const size_t WFRAG = (size_t)E * E;       // 1M elems
  const size_t QKVN  = (size_t)BATCH * HEADS * SEQ * HD; // 8M elems

  char* w = (char*)d_ws;
  size_t off = 0;
  float* twp = (float*)(w + off); off += 4096;
  __bf16* xfh = (__bf16*)(w + off); off += AFRAG * 2;   // reused for ctx frags
  __bf16* xfl = (__bf16*)(w + off); off += AFRAG * 2;
  __bf16* wqh = (__bf16*)(w + off); off += WFRAG * 2;
  __bf16* wql = (__bf16*)(w + off); off += WFRAG * 2;
  __bf16* wkh = (__bf16*)(w + off); off += WFRAG * 2;
  __bf16* wkl = (__bf16*)(w + off); off += WFRAG * 2;
  __bf16* wvh = (__bf16*)(w + off); off += WFRAG * 2;
  __bf16* wvl = (__bf16*)(w + off); off += WFRAG * 2;
  __bf16* woh = (__bf16*)(w + off); off += WFRAG * 2;
  __bf16* wol = (__bf16*)(w + off); off += WFRAG * 2;
  __bf16* Qhb = (__bf16*)(w + off); off += QKVN * 2;
  __bf16* Qlb = (__bf16*)(w + off); off += QKVN * 2;
  __bf16* Khb = (__bf16*)(w + off); off += QKVN * 2;
  __bf16* Klb = (__bf16*)(w + off); off += QKVN * 2;
  __bf16* Vth = (__bf16*)(w + off); off += QKVN * 2;
  __bf16* Vtl = (__bf16*)(w + off); off += QKVN * 2;
  float*  ctx = (float*)(w + off);  off += AFRAG * 4;

  tw_kernel<<<BATCH, 32, 0, stream>>>(time_enc, Wt, bt, twp);

  // pre-split inputs once (kills per-iteration cvt VALU work in the GEMMs)
  afrag_kernel<<<(MT * KT) / 8, 256, 0, stream>>>(x, xfh, xfl);
  bfrag_kernel<<<(KT * NT) / 8, 256, 0, stream>>>(Wq, wqh, wql);
  bfrag_kernel<<<(KT * NT) / 8, 256, 0, stream>>>(Wk, wkh, wkl);
  bfrag_kernel<<<(KT * NT) / 8, 256, 0, stream>>>(Wv, wvh, wvl);
  bfrag_kernel<<<(KT * NT) / 8, 256, 0, stream>>>(Wo, woh, wol);

  qkv_kernel<<<(MT * NT) / 8, 256, 0, stream>>>(xfh, xfl, wqh, wql, wkh, wkl,
                                                wvh, wvl, Qhb, Qlb, Khb, Klb,
                                                Vth, Vtl);

  attn_kernel<<<(BATCH * HEADS * (SEQ / 16)) / 8, 256, 0, stream>>>(
      Qhb, Qlb, Khb, Klb, Vth, Vtl, twp, ctx);

  // re-split ctx into the (now free) x fragment buffers
  afrag_kernel<<<(MT * KT) / 8, 256, 0, stream>>>(ctx, xfh, xfl);

  oproj_kernel<<<(MT * NT) / 8, 256, 0, stream>>>(xfh, xfl, woh, wol, bo, out);
}
